// ElmanRNN_48249662603294
// MI455X (gfx1250) — matrix operand order repaired and checked by reading, not compiled, not run on hardware
//
#include <hip/hip_runtime.h>
#include <hip/hip_bf16.h>
#include <stdint.h>

#define B_ 32
#define S_ 4096
#define I_ 256
#define H_ 512
#define O_ 256
#define NWG 8

typedef __attribute__((ext_vector_type(16))) __bf16 v16bf;
typedef __attribute__((ext_vector_type(8)))  float  v8f;

// fast tanh: native V_TANH_F32 if the builtin exists, else libm fallback
__device__ __forceinline__ float fast_tanh(float v) {
#if __has_builtin(__builtin_amdgcn_tanhf)
    return __builtin_amdgcn_tanhf(v);
#elif __has_builtin(__builtin_amdgcn_tanh_f32)
    return __builtin_amdgcn_tanh_f32(v);
#else
    return tanhf(v);
#endif
}

// ---------------- workspace layout (bytes) ----------------
// counter | x A-frags (bf16) | Wx B-frags | Wh B-frags | Wfc B-frags | h double buffer
#define WS_CNT     0ull
#define WS_XFRAG   256ull
#define XFRAG_ELEMS (33554432ull)                    // S*2mt*8kt*32lane*16slot
#define WS_WX      (WS_XFRAG + XFRAG_ELEMS * 2)      // [32nt][8kt][32][16]
#define WX_ELEMS   (131072ull)
#define WS_WH      (WS_WX + WX_ELEMS * 2)            // [32nt][16kt][32][16]
#define WH_ELEMS   (262144ull)
#define WS_WFC     (WS_WH + WH_ELEMS * 2)            // [16nt][16kt][32][16]
#define WFC_ELEMS  (131072ull)
#define WS_HBUF    (WS_WFC + WFC_ELEMS * 2)          // [2buf][2mt][16kt][32][16]
#define HBUF_ELEMS (32768ull)

// A-matrix 16x32 bf16 layout: lane<16 holds K 0..7,16..23 ; lane>=16 holds 8..15,24..31
__device__ __forceinline__ int a_klocal(int lane, int s) {
    return s + ((s < 8) ? 0 : 8) + ((lane < 16) ? 0 : 8);
}
// B-matrix 32x16 bf16 layout: lane<16 holds K 0..7,16..23 ; lane>=16 holds 8..15,24..31 ; N = lane%16
__device__ __forceinline__ int b_klocal(int lane, int s) {
    return s + ((s < 8) ? 0 : 8) + ((lane < 16) ? 0 : 8);
}

// ---------------- staging: weights -> B-fragment order, zero h state + barrier counter ----
__global__ __launch_bounds__(256) void stage_weights(const float* __restrict__ Wrnn,
                                                     const float* __restrict__ Wfc,
                                                     uint8_t* __restrict__ ws) {
    unsigned long long tid = (unsigned long long)blockIdx.x * 256 + threadIdx.x;
    if (tid == 0) *(volatile int*)(ws + WS_CNT) = 0;
    __bf16* wx  = (__bf16*)(ws + WS_WX);
    __bf16* wh  = (__bf16*)(ws + WS_WH);
    __bf16* wfc = (__bf16*)(ws + WS_WFC);
    __bf16* hb  = (__bf16*)(ws + WS_HBUF);
    if (tid < WX_ELEMS) {
        unsigned e = (unsigned)tid;
        int s = e & 15, lane = (e >> 4) & 31, kt = (e >> 9) & 7, nt = e >> 12;
        int k = kt * 32 + b_klocal(lane, s);
        int n = nt * 16 + (lane & 15);
        wx[e] = (__bf16)Wrnn[k * H_ + n];                       // W_x = rows [0,256)
    } else if (tid < WX_ELEMS + WH_ELEMS) {
        unsigned e = (unsigned)(tid - WX_ELEMS);
        int s = e & 15, lane = (e >> 4) & 31, kt = (e >> 9) & 15, nt = e >> 13;
        int k = kt * 32 + b_klocal(lane, s);
        int n = nt * 16 + (lane & 15);
        wh[e] = (__bf16)Wrnn[(I_ + k) * H_ + n];                // W_h = rows [256,768)
    } else if (tid < WX_ELEMS + WH_ELEMS + WFC_ELEMS) {
        unsigned e = (unsigned)(tid - WX_ELEMS - WH_ELEMS);
        int s = e & 15, lane = (e >> 4) & 31, kt = (e >> 9) & 15, nt = e >> 13;
        int k = kt * 32 + b_klocal(lane, s);
        int n = nt * 16 + (lane & 15);
        wfc[e] = (__bf16)Wfc[k * O_ + n];
    } else if (tid < WX_ELEMS + WH_ELEMS + WFC_ELEMS + HBUF_ELEMS) {
        hb[tid - WX_ELEMS - WH_ELEMS - WFC_ELEMS] = (__bf16)0.0f;  // h0 = 0
    }
}

// ---------------- staging: x -> per-timestep A-fragment order (bf16) ----------------
__global__ __launch_bounds__(256) void stage_x(const float* __restrict__ x,
                                               uint8_t* __restrict__ ws) {
    unsigned long long tid = (unsigned long long)blockIdx.x * 256 + threadIdx.x; // 33,554,432
    __bf16* xf = (__bf16*)(ws + WS_XFRAG);
    int s    = (int)(tid & 15);
    int lane = (int)((tid >> 4) & 31);
    int kt   = (int)((tid >> 9) & 7);
    int mt   = (int)((tid >> 12) & 1);
    int t    = (int)(tid >> 13);
    int m = mt * 16 + (lane & 15);
    int i = kt * 32 + a_klocal(lane, s);
    xf[tid] = (__bf16)x[((unsigned long long)m * S_ + t) * I_ + i];
}

// ---------------- fused recurrent scan + output projection ----------------
__global__ __launch_bounds__(256) void rnn_scan(const float* __restrict__ brnn,
                                                const float* __restrict__ bfc,
                                                float* __restrict__ out,
                                                uint8_t* __restrict__ ws) {
    const int w    = blockIdx.x;            // 0..7 : owns hidden cols [64w, 64w+64)
    const int tid  = threadIdx.x;
    const int wv   = tid >> 5;              // 8 waves
    const int lane = tid & 31;
    const int lrow = lane & 15;
    const int hiH  = (lane >> 4) & 1;

    const __bf16* xfrag = (const __bf16*)(ws + WS_XFRAG);
    const __bf16* gwx   = (const __bf16*)(ws + WS_WX);
    const __bf16* gwh   = (const __bf16*)(ws + WS_WH);
    const __bf16* gwfc  = (const __bf16*)(ws + WS_WFC);
    __bf16* hbuf = (__bf16*)(ws + WS_HBUF);
    int*    cnt  = (int*)(ws + WS_CNT);

    __shared__ alignas(32) __bf16 sWx [4][8][32][16];    // 32 KB
    __shared__ alignas(32) __bf16 sWh [4][16][32][16];   // 64 KB
    __shared__ alignas(32) __bf16 sWfc[2][16][32][16];   // 32 KB
    __shared__ float sBr[64];
    __shared__ float sBf[32];

    { // one-time weight load into LDS (slices are contiguous in staged layout)
        const uint4* s0 = (const uint4*)(gwx + (unsigned long long)w * 16384);
        uint4* d0 = (uint4*)&sWx[0][0][0][0];
        for (int e = tid; e < 2048; e += 256) d0[e] = s0[e];
        const uint4* s1 = (const uint4*)(gwh + (unsigned long long)w * 32768);
        uint4* d1 = (uint4*)&sWh[0][0][0][0];
        for (int e = tid; e < 4096; e += 256) d1[e] = s1[e];
        const uint4* s2 = (const uint4*)(gwfc + (unsigned long long)w * 16384);
        uint4* d2 = (uint4*)&sWfc[0][0][0][0];
        for (int e = tid; e < 2048; e += 256) d2[e] = s2[e];
        if (tid < 64) sBr[tid] = brnn[w * 64 + tid];
        if (tid < 32) sBf[tid] = bfc[w * 32 + tid];
    }
    __syncthreads();

    const int mt   = wv >> 2;                       // h-part: wave -> (M-tile, local N-tile)
    const int ntl  = wv & 3;
    const int ncol = w * 64 + ntl * 16 + lrow;      // global hidden column of this lane
    int epoch = 0;

    for (int t = 0; t < S_; ++t) {
        const int cur = t & 1, nxt = cur ^ 1;
        const float br = sBr[ntl * 16 + lrow];
        v8f acc = (v8f){br, br, br, br, br, br, br, br};

        // x_t @ W_x  (K = 256)
        const __bf16* xa = xfrag + (((unsigned long long)t * 2 + mt) * 8 * 32 + lane) * 16;
        #pragma unroll
        for (int kt = 0; kt < 8; ++kt) {
            v16bf a = *(const v16bf*)(xa + (unsigned long long)kt * 512);
            v16bf b = *(const v16bf*)&sWx[ntl][kt][lane][0];
            acc = __builtin_amdgcn_wmma_f32_16x16x32_bf16(false, a, false, b,
                                                          (short)0, acc, false, false);
        }
        // h_{t-1} @ W_h  (K = 512)
        const __bf16* ha = hbuf + ((cur * 2 + mt) * 16 * 32 + lane) * 16;
        #pragma unroll
        for (int kt = 0; kt < 16; ++kt) {
            v16bf a = *(const v16bf*)(ha + kt * 512);
            v16bf b = *(const v16bf*)&sWh[ntl][kt][lane][0];
            acc = __builtin_amdgcn_wmma_f32_16x16x32_bf16(false, a, false, b,
                                                          (short)0, acc, false, false);
        }
        if (t + 1 < S_)   // pull next timestep's A-fragments toward the WGP
            __builtin_prefetch(xfrag + (((unsigned long long)(t + 1) * 2 + mt) * 8 * 32) * 16, 0, 1);

        // h_new = tanh(.); scatter into next step's A-fragment layout (bf16)
        #pragma unroll
        for (int r = 0; r < 8; ++r) {
            float hv = fast_tanh(acc[r]);
            int mrow  = hiH * 8 + r;                 // row within 16-row tile
            int kl    = ncol & 31;
            int kt2   = ncol >> 5;
            int lane2 = mrow + (((kl >> 3) & 1) ? 16 : 0);
            int s2    = kl - ((kl < 8) ? 0 : ((kl < 24) ? 8 : 16));
            hbuf[(((nxt * 2 + mt) * 16 + kt2) * 32 + lane2) * 16 + s2] = (__bf16)hv;
            if (t == S_ - 1)                          // h_last (fp32) output
                out[(unsigned long long)B_ * S_ * O_ + (mt * 16 + mrow) * H_ + ncol] = hv;
        }

        // ---- one grid barrier per step (monotonic counter, 8 resident WGs) ----
        __threadfence();
        __syncthreads();
        ++epoch;
        if (tid == 0) {
            __hip_atomic_fetch_add(cnt, 1, __ATOMIC_ACQ_REL, __HIP_MEMORY_SCOPE_AGENT);
            while (__hip_atomic_load(cnt, __ATOMIC_ACQUIRE, __HIP_MEMORY_SCOPE_AGENT)
                   < epoch * NWG)
                __builtin_amdgcn_s_sleep(2);
        }
        __syncthreads();
        __threadfence();

        // ---- out_t = h_t @ W_fc + b_fc ; this WG owns out cols [32w, 32w+32) ----
        if (wv < 4) {
            int mto = wv >> 1, nto = wv & 1;
            const float bo = sBf[nto * 16 + lrow];
            v8f oc = (v8f){bo, bo, bo, bo, bo, bo, bo, bo};
            const __bf16* oa = hbuf + ((nxt * 2 + mto) * 16 * 32) * 16;
            #pragma unroll
            for (int kt = 0; kt < 16; ++kt) {
                v16bf a = *(const v16bf*)(oa + (kt * 32 + lane) * 16);
                v16bf b = *(const v16bf*)&sWfc[nto][kt][lane][0];
                oc = __builtin_amdgcn_wmma_f32_16x16x32_bf16(false, a, false, b,
                                                             (short)0, oc, false, false);
            }
            int nout = w * 32 + nto * 16 + lrow;
            #pragma unroll
            for (int r = 0; r < 8; ++r) {
                int m = mto * 16 + hiH * 8 + r;
                out[((unsigned long long)m * S_ + t) * O_ + nout] = oc[r];
            }
        }
    }
}

extern "C" void kernel_launch(void* const* d_in, const int* in_sizes, int n_in,
                              void* d_out, int out_size, void* d_ws, size_t ws_size,
                              hipStream_t stream) {
    const float* x    = (const float*)d_in[0];
    const float* Wrnn = (const float*)d_in[1];
    const float* brnn = (const float*)d_in[2];
    const float* Wfc  = (const float*)d_in[3];
    const float* bfc  = (const float*)d_in[4];
    float*   out = (float*)d_out;
    uint8_t* ws  = (uint8_t*)d_ws;

    stage_weights<<<2176, 256, 0, stream>>>(Wrnn, Wfc, ws);     // 557,056 threads
    stage_x<<<131072, 256, 0, stream>>>(x, ws);                 // 33.5M threads
    rnn_scan<<<NWG, 256, 0, stream>>>(brnn, bfc, out, ws);      // 8 WGs x 8 waves
}
